// CompositionalAttention_33071248180040
// MI455X (gfx1250) — compile-verified, matrix operand order checked by reading, not run on hardware
//
#include <hip/hip_runtime.h>
#include <hip/hip_bf16.h>

// ---------------------------------------------------------------------------
// CDNA5 (gfx1250) compositional attention, bf16 WMMA pipeline.
// Shapes: T=2048, B=2, E=1024, H=16, R=8, HD=64, VD=128, SEL=64.
// Wvq/bvq/bsc provably cancel in the rule softmax (shift invariance) -> dropped.
// ---------------------------------------------------------------------------

typedef __bf16 bf16;
typedef __attribute__((ext_vector_type(16))) __bf16 v16bf;
typedef __attribute__((ext_vector_type(8)))  __bf16 v8bf;
typedef __attribute__((ext_vector_type(8)))  float  v8f;

__device__ __forceinline__ bf16 f2bf(float f) {
    union { float f; unsigned u; } x; x.f = f;
    unsigned r = x.u + 0x7FFFu + ((x.u >> 16) & 1u);   // round-to-nearest-even
    union { unsigned short s; __bf16 b; } y; y.s = (unsigned short)(r >> 16);
    return y.b;
}

__device__ __forceinline__ unsigned short f2bfbits(float f) {
    union { float f; unsigned u; } x; x.f = f;
    unsigned r = x.u + 0x7FFFu + ((x.u >> 16) & 1u);
    return (unsigned short)(r >> 16);
}

__device__ __forceinline__ v8f zero8() {
    v8f z = {0.f, 0.f, 0.f, 0.f, 0.f, 0.f, 0.f, 0.f};
    return z;
}

// Per-lane 16-bit fragment load. CDNA5 16-bit A(16x32) layout per lane:
//   lanes 0-15 : K = {kh+0..kh+7, kh+16..kh+23} with kh=0
//   lanes 16-31: same with kh=8
// Caller passes p = row_base + kh; we load two contiguous 16B chunks.
__device__ __forceinline__ v16bf frag_ld(const bf16* p) {
    v8bf lo = *(const v8bf*)(p);
    v8bf hi = *(const v8bf*)(p + 16);
    v16bf r;
#pragma unroll
    for (int i = 0; i < 8; ++i) { r[i] = lo[i]; r[i + 8] = hi[i]; }
    return r;
}

__device__ __forceinline__ v8f wmma_bf16(v16bf a, v16bf b, v8f c) {
    return __builtin_amdgcn_wmma_f32_16x16x32_bf16(
        /*neg_a=*/false, a, /*neg_b=*/false, b,
        /*c_mod=*/(short)0, c, /*reuse_a=*/false, /*reuse_b=*/false);
}

// ---------------------------------------------------------------------------
// Kernel 0: f32 -> bf16 conversion
// ---------------------------------------------------------------------------
__global__ void cvt_f32_bf16(const float* __restrict__ src, bf16* __restrict__ dst, int n) {
    int i = blockIdx.x * blockDim.x + threadIdx.x;
    if (i < n) dst[i] = f2bf(src[i]);
}

// ---------------------------------------------------------------------------
// Kernel 1: fused q/k/v projections. X:(4096x1024) bf16, W row-major bf16.
// grid = (128, 16, 3), block = 256 (8 waves, each owns one 16x16 out tile).
// Epilogue permutes into attention-friendly layouts:
//   q -> [b,h,t,hd] bf16, scaled by 1/8 ; k -> [b,h,t,hd] bf16
//   v -> transposed [b,r,vd,t] bf16
// ---------------------------------------------------------------------------
__global__ void proj_kernel(const bf16* __restrict__ X,
                            const bf16* __restrict__ Wq, const bf16* __restrict__ Wk,
                            const bf16* __restrict__ Wv,
                            const float* __restrict__ bq, const float* __restrict__ bk,
                            const float* __restrict__ bv,
                            bf16* __restrict__ qbuf, bf16* __restrict__ kbuf,
                            bf16* __restrict__ vbuf) {
    const int lane = threadIdx.x & 31, wv_ = threadIdx.x >> 5;
    const int l16 = lane & 15, hlf = lane >> 4, kh = hlf * 8;
    const int mt = wv_ >> 2, nt = wv_ & 3;
    const int m0 = blockIdx.x * 32 + mt * 16;
    const int n0 = blockIdx.y * 64 + nt * 16;
    const int which = blockIdx.z;

    const bf16* W = (which == 0) ? Wq : ((which == 1) ? Wk : Wv);
    const float* bias = (which == 0) ? bq : ((which == 1) ? bk : bv);

    const bf16* arow = X + (size_t)(m0 + l16) * 1024;
    const bf16* brow = W + (size_t)(n0 + l16) * 1024;

    v8f acc = zero8();
    for (int k0 = 0; k0 < 1024; k0 += 32) {
        __builtin_prefetch(arow + k0 + 64, 0, 0);
        __builtin_prefetch(brow + k0 + 64, 0, 0);
        v16bf a = frag_ld(arow + k0 + kh);
        v16bf b = frag_ld(brow + k0 + kh);
        acc = wmma_bf16(a, b, acc);
    }

#pragma unroll
    for (int r = 0; r < 8; ++r) {
        int row = m0 + r + hlf * 8;       // C layout: VGPR r -> M=r (lanes 0-15), M=r+8 (16-31)
        int col = n0 + l16;
        int t = row >> 1, b = row & 1;    // query rows are (t,b) row-major with B=2
        float val = acc[r] + bias[col];
        if (which == 0) {
            val *= 0.125f;                // HD^-0.5
            int h = col >> 6, hd = col & 63;
            qbuf[((((size_t)b * 16 + h) * 2048 + t) << 6) + hd] = f2bf(val);
        } else if (which == 1) {
            int h = col >> 6, hd = col & 63;
            kbuf[((((size_t)b * 16 + h) * 2048 + t) << 6) + hd] = f2bf(val);
        } else {
            int rr = col >> 7, vd = col & 127;
            vbuf[(((size_t)b * 8 + rr) * 128 + vd) * 2048 + t] = f2bf(val);  // v transposed
        }
    }
}

// ---------------------------------------------------------------------------
// Kernel 2: fused attention + rule selection.
// grid = (T/16=128, B*H=32), block = 256 = 8 waves. 16 query rows per block.
// Phase 1: waves split S for online-softmax stats (m, l).
// Phase 2: per 256-wide S block, wave w computes one 16x32 exp(S-m) chunk
//          (QK^T done once per block!), stages bf16 P in LDS row-major 16x32
//          (natural s order); after barrier every wave (= rule)
//          consumes all 8 chunks against its v^T slice (64 WMMAs / block).
// Epilogue: normalize by l, per-rule score via Wsc[64:192], rule softmax,
//          weighted reduction via ds_add_f32, bf16 store to [t*B+b, h*128+vd].
// ---------------------------------------------------------------------------
__global__ void attn_kernel(const bf16* __restrict__ qbuf, const bf16* __restrict__ kbuf,
                            const bf16* __restrict__ vbuf, const float* __restrict__ Wsc,
                            bf16* __restrict__ hbuf) {
    __shared__ float mpart[8 * 16];
    __shared__ float lpart[8 * 16];
    __shared__ float scoreS[16 * 8];
    __shared__ float obuf[16 * 128];
    __shared__ __align__(16) unsigned short pstage[8 * 16 * 32];  // 8 chunks of 16x32 bf16 P

    const int tid = threadIdx.x;
    const int lane = tid & 31, w = tid >> 5;
    const int l16 = lane & 15, hlf = lane >> 4, kh = hlf * 8;
    const int bh = blockIdx.y;
    const int b = bh >> 4, h = bh & 15;
    const int t0 = blockIdx.x * 16;

    for (int i = tid; i < 16 * 128; i += 256) obuf[i] = 0.f;

    const bf16* qb = qbuf + (size_t)bh * 2048 * 64;
    const bf16* kb = kbuf + (size_t)bh * 2048 * 64;

    // q fragments for this row tile (live across both phases)
    v16bf qa0 = frag_ld(qb + (size_t)(t0 + l16) * 64 + kh);
    v16bf qa1 = frag_ld(qb + (size_t)(t0 + l16) * 64 + 32 + kh);

    // ---- Phase 1: online softmax stats, S split across waves -------------
    float run_m[8], run_l[8];
#pragma unroll
    for (int r = 0; r < 8; ++r) { run_m[r] = -1e30f; run_l[r] = 0.f; }

    for (int s0 = w * 256; s0 < w * 256 + 256; s0 += 16) {
        v16bf k0f = frag_ld(kb + (size_t)(s0 + l16) * 64 + kh);
        v16bf k1f = frag_ld(kb + (size_t)(s0 + l16) * 64 + 32 + kh);
        v8f c = zero8();
        c = wmma_bf16(qa0, k0f, c);
        c = wmma_bf16(qa1, k1f, c);
#pragma unroll
        for (int r = 0; r < 8; ++r) {
            float v = c[r];
            float mx = v;
#pragma unroll
            for (int mk = 1; mk < 16; mk <<= 1) mx = fmaxf(mx, __shfl_xor(mx, mk, 32));
            float nm = fmaxf(run_m[r], mx);
            float p = __expf(v - nm);
#pragma unroll
            for (int mk = 1; mk < 16; mk <<= 1) p += __shfl_xor(p, mk, 32);
            run_l[r] = run_l[r] * __expf(run_m[r] - nm) + p;
            run_m[r] = nm;
        }
    }
    if (l16 == 0) {
#pragma unroll
        for (int r = 0; r < 8; ++r) {
            int row = r + hlf * 8;
            mpart[w * 16 + row] = run_m[r];
            lpart[w * 16 + row] = run_l[r];
        }
    }
    __syncthreads();

    float mrow[8], lrow[8];
#pragma unroll
    for (int r = 0; r < 8; ++r) {
        int row = r + hlf * 8;
        float m = -1e30f;
        for (int ww = 0; ww < 8; ++ww) m = fmaxf(m, mpart[ww * 16 + row]);
        float l = 0.f;
        for (int ww = 0; ww < 8; ++ww) l += lpart[ww * 16 + row] * __expf(mpart[ww * 16 + row] - m);
        mrow[r] = m;
        lrow[r] = l;
    }

    // ---- Phase 2: one rule per wave, QK^T shared through LDS -------------
    const bf16* vb = vbuf + ((size_t)(b * 8 + w)) * 128 * 2048;  // [vd][t]
    v8f acc[8];
#pragma unroll
    for (int nt = 0; nt < 8; ++nt) acc[nt] = zero8();

    for (int sblk = 0; sblk < 2048; sblk += 256) {
        // each wave produces the exp(S-m) chunk for s in [sblk+w*32, +32)
        int scw = sblk + w * 32;
#pragma unroll
        for (int hh = 0; hh < 2; ++hh) {
            int s0 = scw + hh * 16;
            v16bf k0f = frag_ld(kb + (size_t)(s0 + l16) * 64 + kh);
            v16bf k1f = frag_ld(kb + (size_t)(s0 + l16) * 64 + 32 + kh);
            v8f c = zero8();
            c = wmma_bf16(qa0, k0f, c);
            c = wmma_bf16(qa1, k1f, c);
#pragma unroll
            for (int r = 0; r < 8; ++r) {
                int row = r + hlf * 8;
                float p = __expf(c[r] - mrow[r]);
                pstage[w * 512 + row * 32 + hh * 16 + l16] = f2bfbits(p);
            }
        }
        __syncthreads();
        // consume all 8 chunks: 64 WMMAs against this rule's v^T slice
#pragma unroll
        for (int cc = 0; cc < 8; ++cc) {
            v16bf pa = frag_ld((const bf16*)pstage + cc * 512 + l16 * 32 + kh);
            int sv = sblk + cc * 32;
#pragma unroll
            for (int nt = 0; nt < 8; ++nt) {
                v16bf vf = frag_ld(vb + (size_t)(nt * 16 + l16) * 2048 + sv + kh);
                acc[nt] = wmma_bf16(pa, vf, acc[nt]);
            }
        }
        __syncthreads();
    }

    // ---- Epilogue: normalize, rule score, rule softmax, reduce -----------
    float wsc[8];
#pragma unroll
    for (int nt = 0; nt < 8; ++nt) wsc[nt] = Wsc[64 + nt * 16 + l16];

    float sp[8];
#pragma unroll
    for (int r = 0; r < 8; ++r) {
        float inv = 1.f / lrow[r];
        float s = 0.f;
#pragma unroll
        for (int nt = 0; nt < 8; ++nt) {
            float a = acc[nt][r] * inv;
            acc[nt][r] = a;                 // keep normalized attn
            s += a * wsc[nt];
        }
#pragma unroll
        for (int mk = 1; mk < 16; mk <<= 1) s += __shfl_xor(s, mk, 32);
        sp[r] = s;
    }
    if (l16 == 0) {
#pragma unroll
        for (int r = 0; r < 8; ++r) scoreS[(r + hlf * 8) * 8 + w] = sp[r];
    }
    __syncthreads();

#pragma unroll
    for (int r = 0; r < 8; ++r) {
        int row = r + hlf * 8;
        float mx = -1e30f;
        for (int rr = 0; rr < 8; ++rr) mx = fmaxf(mx, scoreS[row * 8 + rr]);
        float den = 0.f;
        for (int rr = 0; rr < 8; ++rr) den += __expf(scoreS[row * 8 + rr] - mx);
        float sel = __expf(sp[r] - mx) / den;
#pragma unroll
        for (int nt = 0; nt < 8; ++nt)
            atomicAdd(&obuf[row * 128 + nt * 16 + l16], acc[nt][r] * sel);
    }
    __syncthreads();

    for (int i = tid; i < 16 * 128; i += 256) {
        int row = i >> 7, col = i & 127;
        int t = t0 + row;
        size_t m = (size_t)t * 2 + b;     // (t,b) row
        hbuf[m * 2048 + h * 128 + col] = f2bf(obuf[i]);
    }
}

// ---------------------------------------------------------------------------
// Kernel 3: out projection. Hb:(4096 x 2048) bf16 @ Wo^T (1024 x 2048) + bo.
// grid = (128, 16), block = 256.
// ---------------------------------------------------------------------------
__global__ void outproj_kernel(const bf16* __restrict__ Hb, const bf16* __restrict__ Wo,
                               const float* __restrict__ bo, float* __restrict__ out) {
    const int lane = threadIdx.x & 31, wv_ = threadIdx.x >> 5;
    const int l16 = lane & 15, hlf = lane >> 4, kh = hlf * 8;
    const int mt = wv_ >> 2, nt = wv_ & 3;
    const int m0 = blockIdx.x * 32 + mt * 16;
    const int n0 = blockIdx.y * 64 + nt * 16;

    const bf16* arow = Hb + (size_t)(m0 + l16) * 2048;
    const bf16* brow = Wo + (size_t)(n0 + l16) * 2048;

    v8f acc = zero8();
    for (int k0 = 0; k0 < 2048; k0 += 32) {
        __builtin_prefetch(arow + k0 + 64, 0, 0);
        __builtin_prefetch(brow + k0 + 64, 0, 0);
        acc = wmma_bf16(frag_ld(arow + k0 + kh), frag_ld(brow + k0 + kh), acc);
    }
#pragma unroll
    for (int r = 0; r < 8; ++r) {
        int row = m0 + r + hlf * 8;
        int col = n0 + l16;
        out[(size_t)row * 1024 + col] = acc[r] + bo[col];
    }
}

// ---------------------------------------------------------------------------
extern "C" void kernel_launch(void* const* d_in, const int* in_sizes, int n_in,
                              void* d_out, int out_size, void* d_ws, size_t ws_size,
                              hipStream_t stream) {
    (void)in_sizes; (void)n_in; (void)out_size; (void)ws_size;

    const float* query = (const float*)d_in[0];
    const float* Wq    = (const float*)d_in[1];
    const float* bq    = (const float*)d_in[2];
    const float* Wk    = (const float*)d_in[3];
    const float* bk    = (const float*)d_in[4];
    const float* Wv    = (const float*)d_in[5];
    const float* bv    = (const float*)d_in[6];
    // d_in[7]=Wvq, d_in[8]=bvq, d_in[10]=bsc: cancel in the rule softmax.
    const float* Wsc   = (const float*)d_in[9];
    const float* Wo    = (const float*)d_in[11];
    const float* bo    = (const float*)d_in[12];
    float* out = (float*)d_out;

    // workspace carve-up (bf16 buffers), all sizes in bytes
    char* ws = (char*)d_ws;
    bf16* Xb   = (bf16*)ws;               ws += (size_t)4096 * 1024 * 2;   // query bf16
    bf16* Wqb  = (bf16*)ws;               ws += (size_t)1024 * 1024 * 2;
    bf16* Wkb  = (bf16*)ws;               ws += (size_t)1024 * 1024 * 2;
    bf16* Wvb  = (bf16*)ws;               ws += (size_t)1024 * 1024 * 2;
    bf16* Wob  = (bf16*)ws;               ws += (size_t)1024 * 2048 * 2;
    bf16* qbuf = (bf16*)ws;               ws += (size_t)2 * 16 * 2048 * 64 * 2;
    bf16* kbuf = (bf16*)ws;               ws += (size_t)2 * 16 * 2048 * 64 * 2;
    bf16* vbuf = (bf16*)ws;               ws += (size_t)2 * 8 * 128 * 2048 * 2;
    bf16* hbuf = (bf16*)ws;               ws += (size_t)4096 * 2048 * 2;

    const dim3 blk(256);
    cvt_f32_bf16<<<dim3((4194304 + 255) / 256), blk, 0, stream>>>(query, Xb, 4194304);
    cvt_f32_bf16<<<dim3((1048576 + 255) / 256), blk, 0, stream>>>(Wq, Wqb, 1048576);
    cvt_f32_bf16<<<dim3((1048576 + 255) / 256), blk, 0, stream>>>(Wk, Wkb, 1048576);
    cvt_f32_bf16<<<dim3((1048576 + 255) / 256), blk, 0, stream>>>(Wv, Wvb, 1048576);
    cvt_f32_bf16<<<dim3((2097152 + 255) / 256), blk, 0, stream>>>(Wo, Wob, 2097152);

    proj_kernel<<<dim3(128, 16, 3), blk, 0, stream>>>(Xb, Wqb, Wkb, Wvb, bq, bk, bv,
                                                      qbuf, kbuf, vbuf);

    attn_kernel<<<dim3(128, 32), blk, 0, stream>>>(qbuf, kbuf, vbuf, Wsc, hbuf);

    outproj_kernel<<<dim3(128, 16), blk, 0, stream>>>(hbuf, Wob, bo, out);
}
